// BSQLinear_13211319403236
// MI455X (gfx1250) — matrix operand order repaired and checked by reading, not compiled, not run on hardware
//
#include <hip/hip_runtime.h>

typedef __attribute__((ext_vector_type(4)))  float  f32x4;
typedef __attribute__((ext_vector_type(8)))  float  v8f;
typedef __attribute__((ext_vector_type(8)))  __bf16 bf16x8;
typedef __attribute__((ext_vector_type(16))) __bf16 bf16x16;

#define IN_F   4096
#define OUT_F  4096
#define M_ROWS 8192

#define BM 128
#define BN 128
#define BK 32
#define PAD 8
#define LDA (BK + PAD)     // 40 bf16 -> 80B LDS row stride, conflict-free b128 frag loads
#define BSTRIDE (BN * LDA) // one B buffer in bf16 elements

static __device__ __forceinline__ bf16x16 combine8(bf16x8 a, bf16x8 b) {
    return __builtin_shufflevector(a, b, 0,1,2,3,4,5,6,7,8,9,10,11,12,13,14,15);
}

// Issue 4 async b128 DMAs: 32B of W_hi and 32B of W_lo into LDS (ASYNCcnt).
// Note: INST_OFFSET applies to BOTH the LDS and global address (ISA 08 §4.4),
// so offset:16 moves the second 16B chunk on both sides.
static __device__ __forceinline__ void async_w(unsigned ldsH, unsigned ldsL,
                                               const __bf16* gh, const __bf16* gl) {
    asm volatile(
        "global_load_async_to_lds_b128 %0, %2, off\n\t"
        "global_load_async_to_lds_b128 %0, %2, off offset:16\n\t"
        "global_load_async_to_lds_b128 %1, %3, off\n\t"
        "global_load_async_to_lds_b128 %1, %3, off offset:16"
        :: "v"(ldsH), "v"(ldsL),
           "v"((unsigned long long)(uintptr_t)gh),
           "v"((unsigned long long)(uintptr_t)gl)
        : "memory");
}

// -------- Kernel A: reconstruct W (4096x4096) and split fp32 -> bf16 hi/lo --------
__global__ __launch_bounds__(256) void bsq_reconstruct_w(
    const float* __restrict__ vq,     // (16384, 4, 32)
    const float* __restrict__ wdec,   // (4, 32, 256)
    const float* __restrict__ bdec,   // (4, 256)
    const float* __restrict__ dmean,  // (4, 1)
    const float* __restrict__ dstd,   // (4, 1)
    __bf16* __restrict__ Whi,         // (4096, 4096)
    __bf16* __restrict__ Wlo)         // (4096, 4096)
{
    const int idx = blockIdx.x * 256 + threadIdx.x;   // over 4096*4096
    const int o = idx >> 12;
    const int i = idx & (IN_F - 1);
    const int p  = o >> 10;
    const int op = o & 1023;
    const int nn = op * 16 + (i >> 8);
    const int c  = i & 255;

    const float* __restrict__ vrow = vq + ((size_t)nn * 4 + p) * 32;
    const float* __restrict__ wcol = wdec + p * 8192 + c;

    float acc = bdec[p * 256 + c];
#pragma unroll
    for (int l = 0; l < 32; ++l)
        acc = fmaf(vrow[l], wcol[l * 256], acc);

    const float s = dstd[p] + 1e-6f;
    const float v = fmaf(acc, s, dmean[p]);
    const __bf16 hi = (__bf16)v;
    const __bf16 lo = (__bf16)(v - (float)hi);
    Whi[idx] = hi;
    Wlo[idx] = lo;
}

// -------- Kernel B: out = x * W^T + bias, bf16x3 split-precision WMMA GEMM --------
// Block tile 128x128x32, 8 waves, wave tile 32x64 (2x4 WMMA tiles).
// Software pipeline: W tiles double-buffered in LDS via async DMA issued one
// stage ahead; x tile prefetched into VGPRs one stage ahead.
__global__ __launch_bounds__(256) void bsq_gemm_wmma(
    const float*  __restrict__ x,     // (8192, 4096) fp32
    const __bf16* __restrict__ Whi,   // (4096, 4096)
    const __bf16* __restrict__ Wlo,   // (4096, 4096)
    const float*  __restrict__ bias,  // (4096,)
    float* __restrict__ out)          // (8192, 4096)
{
    __shared__ __bf16 Ah[BM * LDA];
    __shared__ __bf16 Al[BM * LDA];
    __shared__ __bf16 Bh[2 * BSTRIDE];
    __shared__ __bf16 Bl[2 * BSTRIDE];

    const int t    = threadIdx.x;
    const int lane = t & 31;
    const int wave = t >> 5;
    const int wm   = wave & 3;        // 0..3 -> 32-row slab
    const int wn   = wave >> 2;       // 0..1 -> 64-col slab

    const int mblk = blockIdx.y * BM;
    const int nblk = blockIdx.x * BN;

    const int l15    = lane & 15;
    const int hiHalf = lane >> 4;

    // staging maps: 2 threads per row, 16 elements (x: floats, W: bf16) each
    const int xrow = t >> 1;
    const int xcol = (t & 1) * 16;
    const int wrow = t >> 1;
    const int wcol = (t & 1) * 16;

    const unsigned ldsBh0 = (unsigned)(uintptr_t)&Bh[wrow * LDA + wcol];
    const unsigned ldsBl0 = (unsigned)(uintptr_t)&Bl[wrow * LDA + wcol];
    const unsigned bufB   = (unsigned)(BSTRIDE * sizeof(__bf16));   // 10240 B

    const __bf16* gBh0 = Whi + (size_t)(nblk + wrow) * IN_F + wcol;
    const __bf16* gBl0 = Wlo + (size_t)(nblk + wrow) * IN_F + wcol;
    const float*  xbase = x + (size_t)(mblk + xrow) * IN_F + xcol;

    v8f acc[2][4];
#pragma unroll
    for (int mi = 0; mi < 2; ++mi)
#pragma unroll
        for (int ni = 0; ni < 4; ++ni)
            acc[mi][ni] = (v8f)(0.0f);

    // ---- pipeline prologue: stage 0 ----
    async_w(ldsBh0, ldsBl0, gBh0, gBl0);          // W(0) -> B buffer 0
    f32x4 xr[4];
#pragma unroll
    for (int j = 0; j < 4; ++j)
        xr[j] = *(const f32x4*)(xbase + j * 4);   // x(0) -> regs

    for (int kb = 0, it = 0; kb < IN_F; kb += BK, ++it) {
        const int cur = it & 1;
        const int kn  = kb + BK;

        __syncthreads();   // all waves done computing previous stage (A reuse safe)

        // ---- convert prefetched x regs -> bf16 hi/lo in LDS ----
#pragma unroll
        for (int j = 0; j < 2; ++j) {
            const f32x4 a = xr[2 * j], b2 = xr[2 * j + 1];
            float f[8] = {a.x, a.y, a.z, a.w, b2.x, b2.y, b2.z, b2.w};
            bf16x8 h, l;
#pragma unroll
            for (int e = 0; e < 8; ++e) {
                const __bf16 hh = (__bf16)f[e];
                h[e] = hh;
                l[e] = (__bf16)(f[e] - (float)hh);
            }
            const int off = xrow * LDA + xcol + j * 8;
            *(bf16x8*)&Ah[off] = h;
            *(bf16x8*)&Al[off] = l;
        }

        // ---- prefetch next x tile into regs (hides behind wait+compute) ----
        if (kn < IN_F) {
#pragma unroll
            for (int j = 0; j < 4; ++j)
                xr[j] = *(const f32x4*)(xbase + kn + j * 4);
        }

        asm volatile("s_wait_asynccnt 0x0" ::: "memory");   // W(cur) resident
        __syncthreads();

        // ---- issue W(next) async DMA into the other B buffer ----
        if (kn < IN_F) {
            const unsigned boff = cur ? 0u : bufB;          // next buffer
            async_w(ldsBh0 + boff, ldsBl0 + boff, gBh0 + kn, gBl0 + kn);
        }

        // ---- A fragments (persist across ni) ----
        bf16x16 afh[2], afl[2];
#pragma unroll
        for (int mi = 0; mi < 2; ++mi) {
            const int row = (wm * 32 + mi * 16 + l15) * LDA;
            afh[mi] = combine8(*(const bf16x8*)&Ah[row + hiHalf * 8],
                               *(const bf16x8*)&Ah[row + 16 + hiHalf * 8]);
            afl[mi] = combine8(*(const bf16x8*)&Al[row + hiHalf * 8],
                               *(const bf16x8*)&Al[row + 16 + hiHalf * 8]);
        }

        // ---- B fragments loaded per-ni, consumed immediately ----
        const __bf16* __restrict__ BhC = &Bh[cur * BSTRIDE];
        const __bf16* __restrict__ BlC = &Bl[cur * BSTRIDE];
#pragma unroll
        for (int ni = 0; ni < 4; ++ni) {
            const int row = (wn * 64 + ni * 16 + l15) * LDA + hiHalf * 8;
            bf16x16 bh = combine8(*(const bf16x8*)&BhC[row],
                                  *(const bf16x8*)&BhC[row + 16]);
            bf16x16 bl = combine8(*(const bf16x8*)&BlC[row],
                                  *(const bf16x8*)&BlC[row + 16]);
#pragma unroll
            for (int mi = 0; mi < 2; ++mi) {
                acc[mi][ni] = __builtin_amdgcn_wmma_f32_16x16x32_bf16(
                    false, afh[mi], false, bh, (short)0, acc[mi][ni], false, false);
                acc[mi][ni] = __builtin_amdgcn_wmma_f32_16x16x32_bf16(
                    false, afh[mi], false, bl, (short)0, acc[mi][ni], false, false);
                acc[mi][ni] = __builtin_amdgcn_wmma_f32_16x16x32_bf16(
                    false, afl[mi], false, bh, (short)0, acc[mi][ni], false, false);
            }
        }
    }

    // ---- epilogue: add bias, store fp32 ----
#pragma unroll
    for (int ni = 0; ni < 4; ++ni) {
        const int col = nblk + wn * 64 + ni * 16 + l15;
        const float b = bias[col];
#pragma unroll
        for (int mi = 0; mi < 2; ++mi) {
            const int rbase = mblk + wm * 32 + mi * 16 + hiHalf * 8;
#pragma unroll
            for (int r = 0; r < 8; ++r) {
                out[(size_t)(rbase + r) * OUT_F + col] = acc[mi][ni][r] + b;
            }
        }
    }
}

extern "C" void kernel_launch(void* const* d_in, const int* in_sizes, int n_in,
                              void* d_out, int out_size, void* d_ws, size_t ws_size,
                              hipStream_t stream) {
    const float* x     = (const float*)d_in[0];
    const float* vq    = (const float*)d_in[1];
    const float* wdec  = (const float*)d_in[2];
    const float* bdec  = (const float*)d_in[3];
    const float* dmean = (const float*)d_in[4];
    const float* dstd  = (const float*)d_in[5];
    const float* bias  = (const float*)d_in[6];
    float* out = (float*)d_out;

    __bf16* Whi = (__bf16*)d_ws;
    __bf16* Wlo = Whi + (size_t)OUT_F * IN_F;

    // Stage 1: reconstruct weight, split into bf16 hi/lo (64 MB of workspace)
    bsq_reconstruct_w<<<(OUT_F * IN_F) / 256, 256, 0, stream>>>(
        vq, wdec, bdec, dmean, dstd, Whi, Wlo);

    // Stage 2: pipelined split-precision WMMA GEMM with async-to-LDS W staging
    dim3 grid(OUT_F / BN, M_ROWS / BM);   // (32, 64)
    bsq_gemm_wmma<<<grid, 256, 0, stream>>>(x, Whi, Wlo, bias, out);
}
